// GlobalFeatureExtractorGNN_35871566856973
// MI455X (gfx1250) — matrix operand order repaired and checked by reading, not compiled, not run on hardware
//
#include <hip/hip_runtime.h>
#include <hip/hip_bf16.h>

#define C 128
#define NEG_SLOPE 0.2f

typedef __attribute__((ext_vector_type(16))) __bf16 v16bf;
typedef __attribute__((ext_vector_type(8)))  __bf16 v8bf;
typedef __attribute__((ext_vector_type(8)))  float  v8f;

// ---------- helpers ----------
__device__ __forceinline__ unsigned encodeF(float f) {
  unsigned b = __float_as_uint(f);
  return (b & 0x80000000u) ? ~b : (b | 0x80000000u);
}
__device__ __forceinline__ float decodeF(unsigned k) {
  unsigned b = (k & 0x80000000u) ? (k & 0x7FFFFFFFu) : ~k;
  return __uint_as_float(b);
}
__device__ __forceinline__ float lrelu(float x) {
  return (x > 0.f) ? x : NEG_SLOPE * x;
}

// ---------- elementwise / init ----------
__global__ void k_zero_u32(unsigned* __restrict__ p, int n) {
  int i = blockIdx.x * blockDim.x + threadIdx.x;
  if (i < n) p[i] = 0u;
}

__global__ void k_f32_to_bf16(const float* __restrict__ x, __bf16* __restrict__ xb, int n) {
  int i = blockIdx.x * blockDim.x + threadIdx.x;
  if (i < n) xb[i] = (__bf16)x[i];
}

// Pack W (C x C row-major f32) into per-lane WMMA-B vectors (bf16).
// Layout: Wp[((nt*4 + kt)*32 + lane)*16 + j]
//   lane<16 : b[j] = W[kt*32 +     j + (j<8?0:8)][nt*16 + lane]
//   lane>=16: b[j] = W[kt*32 + 8 + j + (j<8?0:8)][nt*16 + lane-16]
__global__ void k_pack_w(const float* __restrict__ W, __bf16* __restrict__ Wp) {
  int tid = blockIdx.x * blockDim.x + threadIdx.x;   // 0 .. 16383
  int j    = tid & 15;
  int lane = (tid >> 4) & 31;
  int kt   = (tid >> 9) & 3;
  int nt   = tid >> 11;
  int nn = nt * 16 + (lane & 15);
  int kk = kt * 32 + ((lane < 16) ? 0 : 8) + j + ((j < 8) ? 0 : 8);
  Wp[tid] = (__bf16)W[kk * C + nn];
}

// ---------- WMMA GEMM: h[N x 128] = xb[N x 128] @ W[128 x 128] ----------
// block = (32, 8): 8 waves share one 16x128 A tile (staged via LDS once);
// wave `nt` computes the 16x16 tile at cols [nt*16, nt*16+16).
// K = 128 -> 4 x v_wmma_f32_16x16x32_bf16 per wave.
#define ATILE_STRIDE 136   // 128 + 8 bf16 pad: row start advances 68 words -> banks 4m mod 64
__global__ void k_gemm_wmma(const __bf16* __restrict__ xb, const __bf16* __restrict__ Wp,
                            float* __restrict__ h, int n) {
  __shared__ __align__(16) __bf16 Atile[16 * ATILE_STRIDE];
  int rowTile = blockIdx.x;
  int lane = threadIdx.x;
  int nt   = threadIdx.y;

  // cooperative stage: 256 threads x 8 bf16 (b128) = full 16x128 tile
  {
    int t  = nt * 32 + lane;       // 0..255
    int r  = t >> 4;               // row 0..15
    int k8 = (t & 15) << 3;        // k offset 0,8,...,120
    int gr = rowTile * 16 + r;
    if (gr >= n) gr = n - 1;       // clamp (keeps waves convergent)
    *(v8bf*)(&Atile[r * ATILE_STRIDE + k8]) = *(const v8bf*)(xb + (size_t)gr * C + k8);
  }
  __syncthreads();

  int m  = lane & 15;
  int kb = (lane < 16) ? 0 : 8;
  const __bf16* arow = &Atile[m * ATILE_STRIDE + kb];

  v8f acc = {};
#pragma unroll
  for (int kt = 0; kt < 4; ++kt) {
    v8bf lo = *(const v8bf*)(arow + kt * 32);        // K = kb .. kb+7
    v8bf hi = *(const v8bf*)(arow + kt * 32 + 16);   // K = kb+16 .. kb+23
    v16bf a;
#pragma unroll
    for (int j = 0; j < 8; ++j) { a[j] = lo[j]; a[8 + j] = hi[j]; }
    v16bf b = *(const v16bf*)(Wp + (((nt * 4 + kt) * 32 + lane) << 4));
    acc = __builtin_amdgcn_wmma_f32_16x16x32_bf16(
        /*neg_a=*/false, a, /*neg_b=*/false, b,
        /*c_mod=*/(short)0, acc, /*reuse_a=*/false, /*reuse_b=*/false);
  }

  int col = nt * 16 + m;
  int r0  = rowTile * 16 + ((lane < 16) ? 0 : 8);
  if (rowTile * 16 + 16 <= n) {          // uniform fast path: straight-line stores
#pragma unroll
    for (int r = 0; r < 8; ++r)
      h[(size_t)(r0 + r) * C + col] = acc[r];
  } else {
#pragma unroll
    for (int r = 0; r < 8; ++r)
      if (r0 + r < n) h[(size_t)(r0 + r) * C + col] = acc[r];
  }
}

// ---------- per-node attention logits: als = h@a_src, ald = h@a_dst ----------
// one wave (32 lanes) per node; lane handles 4 channels.
__global__ void k_alpha(const float* __restrict__ h, const float* __restrict__ asrc,
                        const float* __restrict__ adst, float* __restrict__ als,
                        float* __restrict__ ald, int n) {
  int node = blockIdx.x * blockDim.y + threadIdx.y;
  if (node >= n) return;
  int lane = threadIdx.x;
  const float4 hv = *(const float4*)(h + (size_t)node * C + lane * 4);
  const float4 sv = *(const float4*)(asrc + lane * 4);
  const float4 dv = *(const float4*)(adst + lane * 4);
  float s = hv.x * sv.x + hv.y * sv.y + hv.z * sv.z + hv.w * sv.w;
  float d = hv.x * dv.x + hv.y * dv.y + hv.z * dv.z + hv.w * dv.w;
#pragma unroll
  for (int off = 16; off > 0; off >>= 1) {
    s += __shfl_down(s, off, 32);
    d += __shfl_down(d, off, 32);
  }
  if (lane == 0) { als[node] = s; ald[node] = d; }
}

// ---------- edge passes (edges 0..E-1 from edge_index, E..E+N-1 are self loops) ----------
__global__ void k_edge_max(const int* __restrict__ ei, const float* __restrict__ als,
                           const float* __restrict__ ald, unsigned* __restrict__ mEnc,
                           int Eo, int total) {
  int e = blockIdx.x * blockDim.x + threadIdx.x;
  if (e >= total) return;
  int s  = (e < Eo) ? ei[e]      : (e - Eo);
  int dv = (e < Eo) ? ei[Eo + e] : (e - Eo);
  float x = lrelu(als[s] + ald[dv]);
  atomicMax(mEnc + dv, encodeF(x));
}

__global__ void k_edge_denom(const int* __restrict__ ei, const float* __restrict__ als,
                             const float* __restrict__ ald, const unsigned* __restrict__ mEnc,
                             float* __restrict__ denom, int Eo, int total) {
  int e = blockIdx.x * blockDim.x + threadIdx.x;
  if (e >= total) return;
  int s  = (e < Eo) ? ei[e]      : (e - Eo);
  int dv = (e < Eo) ? ei[Eo + e] : (e - Eo);
  float x = lrelu(als[s] + ald[dv]);
  atomicAdd(denom + dv, __expf(x - decodeF(mEnc[dv])));
}

// one wave per edge; lane covers channels lane, lane+32, lane+64, lane+96
__global__ void k_aggregate(const int* __restrict__ ei, const float* __restrict__ als,
                            const float* __restrict__ ald, const unsigned* __restrict__ mEnc,
                            const float* __restrict__ denom, const float* __restrict__ h,
                            float* __restrict__ agg, int Eo, int total) {
  int e = blockIdx.x * blockDim.y + threadIdx.y;
  if (e >= total) return;
  int lane = threadIdx.x;
  int s  = (e < Eo) ? ei[e]      : (e - Eo);
  int dv = (e < Eo) ? ei[Eo + e] : (e - Eo);
  float x = lrelu(als[s] + ald[dv]);
  float a = __expf(x - decodeF(mEnc[dv])) / denom[dv];
  const float* hs = h + (size_t)s * C;
  float* ag = agg + (size_t)dv * C;
#pragma unroll
  for (int c = 0; c < 4; ++c)
    atomicAdd(ag + lane + c * 32, a * hs[lane + c * 32]);
}

// ---------- epilogue: +bias, SELU (in place) ----------
__global__ void k_bias_selu(float* __restrict__ y, const float* __restrict__ b, int n128) {
  int i = blockIdx.x * blockDim.x + threadIdx.x;
  if (i >= n128) return;
  const float SC = 1.0507009873554805f, AL = 1.6732632423543772f;
  float v = y[i] + b[i & (C - 1)];
  y[i] = (v > 0.f) ? SC * v : SC * AL * (__expf(v) - 1.f);
}

// ---------- global mean pool ----------
__global__ void k_pool_cnt(const int* __restrict__ batch, float* __restrict__ cnt, int n) {
  int i = blockIdx.x * blockDim.x + threadIdx.x;
  if (i < n) atomicAdd(cnt + batch[i], 1.f);
}
__global__ void k_pool_acc(const float* __restrict__ y, const int* __restrict__ batch,
                           float* __restrict__ ps, int n128) {
  int i = blockIdx.x * blockDim.x + threadIdx.x;
  if (i >= n128) return;
  atomicAdd(ps + (size_t)batch[i >> 7] * C + (i & (C - 1)), y[i]);
}
__global__ void k_pool_final(const float* __restrict__ ps, const float* __restrict__ cnt,
                             float* __restrict__ out, int g128) {
  int i = blockIdx.x * blockDim.x + threadIdx.x;
  if (i < g128) out[i] = ps[i] / fmaxf(cnt[i >> 7], 1.f);
}

// ---------- launch ----------
extern "C" void kernel_launch(void* const* d_in, const int* in_sizes, int n_in,
                              void* d_out, int out_size, void* d_ws, size_t ws_size,
                              hipStream_t stream) {
  (void)n_in; (void)ws_size;
  const float* u   = (const float*)d_in[0];
  const float* W[2]    = {(const float*)d_in[1], (const float*)d_in[5]};
  const float* asv[2]  = {(const float*)d_in[2], (const float*)d_in[6]};
  const float* adv[2]  = {(const float*)d_in[3], (const float*)d_in[7]};
  const float* bv[2]   = {(const float*)d_in[4], (const float*)d_in[8]};
  const int* ei    = (const int*)d_in[9];
  const int* batch = (const int*)d_in[10];

  const int N  = in_sizes[0] / C;
  const int E  = in_sizes[9] / 2;
  const int G  = out_size / C;
  const int NT = N + E;                 // edges incl. self loops

  // workspace carve (256B aligned)
  char* w = (char*)d_ws;
  size_t off = 0;
  auto carve = [&](size_t bytes) -> void* {
    void* p = (void*)(w + off);
    off = (off + bytes + 255) & ~(size_t)255;
    return p;
  };
  float*    bufA  = (float*)   carve((size_t)N * C * 4);  // agg / layer output
  float*    bufH  = (float*)   carve((size_t)N * C * 4);  // h = x@W
  __bf16*   xb    = (__bf16*)  carve((size_t)N * C * 2);  // bf16 input copy
  __bf16*   Wp    = (__bf16*)  carve((size_t)C * C * 2);  // packed W
  float*    als   = (float*)   carve((size_t)N * 4);
  float*    ald   = (float*)   carve((size_t)N * 4);
  unsigned* mEnc  = (unsigned*)carve((size_t)N * 4);
  float*    denom = (float*)   carve((size_t)N * 4);
  float*    ps    = (float*)   carve((size_t)G * C * 4);
  float*    cnt   = (float*)   carve((size_t)G * 4);

  const int T = 256;
  dim3 wblk(32, 8);
  const float* xin = u;

  for (int l = 0; l < 2; ++l) {
    k_f32_to_bf16<<<(N * C + T - 1) / T, T, 0, stream>>>(xin, xb, N * C);
    k_pack_w<<<(C * C + T - 1) / T, T, 0, stream>>>(W[l], Wp);
    k_gemm_wmma<<<(N + 15) / 16, wblk, 0, stream>>>(xb, Wp, bufH, N);
    k_alpha<<<(N + 7) / 8, wblk, 0, stream>>>(bufH, asv[l], adv[l], als, ald, N);
    k_zero_u32<<<(N + T - 1) / T, T, 0, stream>>>(mEnc, N);
    k_zero_u32<<<(N + T - 1) / T, T, 0, stream>>>((unsigned*)denom, N);
    k_zero_u32<<<(N * C + T - 1) / T, T, 0, stream>>>((unsigned*)bufA, N * C);
    k_edge_max<<<(NT + T - 1) / T, T, 0, stream>>>(ei, als, ald, mEnc, E, NT);
    k_edge_denom<<<(NT + T - 1) / T, T, 0, stream>>>(ei, als, ald, mEnc, denom, E, NT);
    k_aggregate<<<(NT + 7) / 8, wblk, 0, stream>>>(ei, als, ald, mEnc, denom, bufH, bufA, E, NT);
    k_bias_selu<<<(N * C + T - 1) / T, T, 0, stream>>>(bufA, bv[l], N * C);
    xin = bufA;
  }

  k_zero_u32<<<(G * C + T - 1) / T, T, 0, stream>>>((unsigned*)ps, G * C);
  k_zero_u32<<<(G + T - 1) / T, T, 0, stream>>>((unsigned*)cnt, G);
  k_pool_cnt<<<(N + T - 1) / T, T, 0, stream>>>(batch, cnt, N);
  k_pool_acc<<<(N * C + T - 1) / T, T, 0, stream>>>(bufA, batch, ps, N * C);
  k_pool_final<<<(G * C + T - 1) / T, T, 0, stream>>>(ps, cnt, (float*)d_out, G * C);
}
